// GestureRecognitionModel_41721312314276
// MI455X (gfx1250) — matrix operand order repaired and checked by reading, compiled, not run on hardware
//
#include <hip/hip_runtime.h>
#include <hip/hip_bf16.h>
#include <math.h>

// ---------------------------------------------------------------------------
// Gesture LSTM for MI455X (gfx1250): bf16 WMMA recurrence, weights stationary
// in VGPRs, per-step state (x_t, h, c, gates) resident in LDS, and x_t staged
// one timestep ahead via async global->LDS copies (ASYNCcnt).
// ---------------------------------------------------------------------------

typedef __attribute__((ext_vector_type(16))) __bf16 v16bf;
typedef __attribute__((ext_vector_type(8)))  __bf16 v8bf;
typedef __attribute__((ext_vector_type(8)))  float  v8f;

#define T_STEPS 512
#define HID     100
#define BATCH   256
#define GATES   (4 * HID)   // 400 = 25 tiles of 16
#define BN_EPS  1e-5f

// ---------------------------------------------------------------------------
// One LSTM layer. Grid = 16 blocks (batch tiles of 16 rows).
// Block = 160 threads = 5 waves; wave w owns gate N-tiles [5w, 5w+5).
// DIN: per-step input width (5 for layer 0, 100 otherwise).
// XF32: layer-0 reads fp32 x from d_in; later layers read bf16 sequence.
// ---------------------------------------------------------------------------
template <int DIN, bool XF32>
__global__ __launch_bounds__(160, 1) void lstm_layer_kernel(
    const float*  __restrict__ xf,        // [B,T,DIN] fp32 (XF32 only)
    const __bf16* __restrict__ xh,        // [B,T,DIN] bf16 (!XF32 only)
    const float*  __restrict__ W_ih,      // [4H, DIN]
    const float*  __restrict__ W_hh,      // [4H, H]
    const float*  __restrict__ bias,      // [4H]
    __bf16*       __restrict__ h_seq_out, // [B,T,H] bf16 or nullptr
    float*        __restrict__ h_last_out)// [B,H] fp32 or nullptr
{
    constexpr int KTOT   = DIN + HID;          // logical K (105 or 200)
    constexpr int NKT    = (KTOT + 31) / 32;   // K-steps of 32 (4 or 7)
    constexpr int KP     = NKT * 32;           // padded K (128 or 224)
    constexpr int RW     = XF32 ? DIN : DIN / 2;   // dwords per (row,t) in x
    constexpr int DWORDS = 16 * RW;                // dwords per x_t tile (80/800)
    constexpr int NL     = (DWORDS + 159) / 160;   // async loads per thread (1/5)

    __shared__ __bf16   sA[16][KP];        // A = [x_t | h], bf16, zero padded
    __shared__ float    sGates[16][GATES]; // f32 gate pre-activations
    __shared__ float    sC[16][HID];       // cell state
    __shared__ float    sBias[GATES];
    __shared__ unsigned sXstage[2][DWORDS];// async x_t staging (double buffer)

    const int tid  = threadIdx.x;
    const int wave = tid >> 5;
    const int lane = tid & 31;
    const int bt   = blockIdx.x;           // batch tile (16 rows)

    const unsigned long long gbase =
        (unsigned long long)(uintptr_t)(XF32 ? (const void*)xf : (const void*)xh);

    // issue one timestep's x tile as async global->LDS B32 copies (GVS mode:
    // saddr = uniform base, vaddr = per-lane 32-bit byte offset).
    auto issue_xtile = [&](int t, int buf) {
#pragma unroll
        for (int q = 0; q < NL; ++q) {
            const int d = tid + q * 160;
            if (d < DWORDS) {
                const int r = d / RW, j2 = d % RW;
                const unsigned voff =
                    (((unsigned)(bt * 16 + r) * T_STEPS + (unsigned)t) * RW + j2) * 4u;
                const unsigned ldsa = (unsigned)(uintptr_t)&sXstage[buf][d];
                asm volatile("global_load_async_to_lds_b32 %0, %1, %2"
                             :: "v"(ldsa), "v"(voff), "s"(gbase)
                             : "memory");
            }
        }
    };

    // ---- init LDS: zero A (covers K padding) and c; stage bias ------------
    for (int i = tid; i < 16 * KP;  i += 160) (&sA[0][0])[i] = (__bf16)0.0f;
    for (int i = tid; i < 16 * HID; i += 160) (&sC[0][0])[i] = 0.0f;
    for (int i = tid; i < GATES;    i += 160) sBias[i] = bias[i];

    // kick off the t=0 x tile while we build the weight fragments
    issue_xtile(0, 0);

    // ---- stationary B fragments (concatenated [W_ih;W_hh]^T, bf16) --------
    // B (32x16, KxN) lane layout: N = lane&15; half-wave selects K base
    // (lanes 0-15 -> kb=0, lanes 16-31 -> kb=8); element e -> K=kb+e (e<8), 16+kb+e-8 (e>=8).
    v16bf Bfrag[NKT][5];
    {
        const int ncol_lo = lane & 15;
        const int kb      = (lane < 16) ? 0 : 8;
#pragma unroll
        for (int ks = 0; ks < NKT; ++ks) {
#pragma unroll
            for (int n = 0; n < 5; ++n) {
                const int ncol = (wave * 5 + n) * 16 + ncol_lo; // gate column
#pragma unroll
                for (int e = 0; e < 16; ++e) {
                    const int k = ks * 32 + kb + (e < 8 ? e : e + 8);
                    float w = 0.0f;
                    if (k < DIN)       w = W_ih[ncol * DIN + k];
                    else if (k < KTOT) w = W_hh[ncol * HID + (k - DIN)];
                    Bfrag[ks][n][e] = (__bf16)w;
                }
            }
        }
    }
    __syncthreads();

    // ---- recurrence over T -------------------------------------------------
    for (int t = 0; t < T_STEPS; ++t) {
        // Phase 0: issue async x tile for t+1; drain this step's tile.
        // ASYNCcnt completes in order, so count <= NL implies step-t loads done.
        if (t + 1 < T_STEPS) {
            issue_xtile(t + 1, (t + 1) & 1);
            asm volatile("s_wait_asynccnt %0" :: "i"(NL) : "memory");
        } else {
            asm volatile("s_wait_asynccnt 0" ::: "memory");
        }
        __syncthreads();   // publish all waves' stage writes

        // Phase 1: repack staged x_t into sA[:, 0:DIN) (pure LDS traffic)
        if constexpr (XF32) {
            for (int d = tid; d < DWORDS; d += 160) {
                const int r = d / RW, c = d % RW;
                sA[r][c] = (__bf16)__uint_as_float(sXstage[t & 1][d]);
            }
        } else {
            for (int d = tid; d < DWORDS; d += 160) {
                const int r = d / RW, c = d % RW;
                *(unsigned*)&sA[r][c * 2] = sXstage[t & 1][d];
            }
        }
        __syncthreads();

        // Phase 2: gates = [x_t,h] @ Bfrag  (bf16 WMMA, f32 accumulate)
        v8f acc[5] = {};
        {
            // A (16x32) lane layout: row = lane&15; lanes>=16 take K offset +8
            const int row  = lane & 15;
            const int koff = (lane < 16) ? 0 : 8;
#pragma unroll
            for (int ks = 0; ks < NKT; ++ks) {
                const v8bf lo = *(const v8bf*)&sA[row][ks * 32 + koff];
                const v8bf hi = *(const v8bf*)&sA[row][ks * 32 + 16 + koff];
                const v16bf a = __builtin_shufflevector(
                    lo, hi, 0, 1, 2, 3, 4, 5, 6, 7, 8, 9, 10, 11, 12, 13, 14, 15);
#pragma unroll
                for (int n = 0; n < 5; ++n) {
                    acc[n] = __builtin_amdgcn_wmma_f32_16x16x32_bf16(
                        false, a, false, Bfrag[ks][n],
                        (short)0, acc[n], false, false);
                }
            }
        }
        // Phase 3: C/D layout -> LDS (VGPR r: M = r + 8*(lane>=16), N=lane&15)
        {
            const int crow = (lane < 16) ? 0 : 8;
#pragma unroll
            for (int n = 0; n < 5; ++n) {
                const int ncol = (wave * 5 + n) * 16 + (lane & 15);
#pragma unroll
                for (int r = 0; r < 8; ++r)
                    sGates[crow + r][ncol] = acc[n][r];
            }
        }
        __syncthreads();

        // Phase 4: pointwise LSTM cell; h back into sA, stream h to global
        for (int i = tid; i < 16 * HID; i += 160) {
            const int r = i / HID, j = i % HID;
            const float gi = sGates[r][j]           + sBias[j];
            const float gf = sGates[r][HID + j]     + sBias[HID + j];
            const float gg = sGates[r][2 * HID + j] + sBias[2 * HID + j];
            const float go = sGates[r][3 * HID + j] + sBias[3 * HID + j];
            const float iv = 1.0f / (1.0f + __expf(-gi));
            const float fv = 1.0f / (1.0f + __expf(-gf));
            const float gv = tanhf(gg);
            const float ov = 1.0f / (1.0f + __expf(-go));
            const float c  = fv * sC[r][j] + iv * gv;
            sC[r][j] = c;
            const float h = ov * tanhf(c);
            sA[r][DIN + j] = (__bf16)h;
            if (h_seq_out)
                h_seq_out[((long long)(bt * 16 + r) * T_STEPS + t) * HID + j] = (__bf16)h;
            if (h_last_out && t == T_STEPS - 1)
                h_last_out[(bt * 16 + r) * HID + j] = h;
        }
        __syncthreads();
    }
}

// ---------------------------------------------------------------------------
// Head: inference BN then 3 tiny dense layers (scalar; ~10 MFLOP total).
// ---------------------------------------------------------------------------
__global__ void bn_kernel(const float* __restrict__ h, const float* __restrict__ g,
                          const float* __restrict__ b, const float* __restrict__ m,
                          const float* __restrict__ v, float* __restrict__ out)
{
    const int id = blockIdx.x * blockDim.x + threadIdx.x;
    if (id >= BATCH * HID) return;
    const int j = id % HID;
    out[id] = (h[id] - m[j]) * rsqrtf(v[j] + BN_EPS) * g[j] + b[j];
}

__global__ void fc_kernel(const float* __restrict__ in, const float* __restrict__ W,
                          const float* __restrict__ bias, float* __restrict__ out,
                          int K, int N, int do_relu)
{
    const int id = blockIdx.x * blockDim.x + threadIdx.x;
    if (id >= BATCH * N) return;
    const int r = id / N, n = id % N;
    float s = bias[n];
    for (int k = 0; k < K; ++k) s += in[r * K + k] * W[n * K + k];
    out[id] = do_relu ? fmaxf(s, 0.0f) : s;
}

// ---------------------------------------------------------------------------
extern "C" void kernel_launch(void* const* d_in, const int* in_sizes, int n_in,
                              void* d_out, int out_size, void* d_ws, size_t ws_size,
                              hipStream_t stream)
{
    const float* x     = (const float*)d_in[0];   // [B,T,5]
    const float* W_ih0 = (const float*)d_in[1];   // [400,5]
    const float* W_hh0 = (const float*)d_in[2];   // [400,100]
    const float* b0    = (const float*)d_in[3];   // [400]
    const float* W_ih  = (const float*)d_in[4];   // [4,400,100]
    const float* W_hh  = (const float*)d_in[5];   // [4,400,100]
    const float* bseq  = (const float*)d_in[6];   // [4,400]
    const float* gamma = (const float*)d_in[7];
    const float* beta  = (const float*)d_in[8];
    const float* rmean = (const float*)d_in[9];
    const float* rvar  = (const float*)d_in[10];
    const float* W1    = (const float*)d_in[11];
    const float* b1    = (const float*)d_in[12];
    const float* W2    = (const float*)d_in[13];
    const float* b2    = (const float*)d_in[14];
    const float* W3    = (const float*)d_in[15];
    const float* b3    = (const float*)d_in[16];

    // workspace layout
    const size_t SEQ_BYTES = (size_t)BATCH * T_STEPS * HID * sizeof(__bf16); // 26.2 MB
    char*   ws    = (char*)d_ws;
    __bf16* seqA  = (__bf16*)ws;
    __bf16* seqB  = (__bf16*)(ws + SEQ_BYTES);
    float*  hlast = (float*)(ws + 2 * SEQ_BYTES);
    float*  xn    = hlast + BATCH * HID;
    float*  y1    = xn    + BATCH * HID;
    float*  y2    = y1    + BATCH * HID;

    const dim3 grid(BATCH / 16), blk(160);
    const int WL = GATES * HID; // per-layer weight stride for W_ih/W_hh stacks

    // layer 0: fp32 x -> seqA
    lstm_layer_kernel<5, true><<<grid, blk, 0, stream>>>(
        x, nullptr, W_ih0, W_hh0, b0, seqA, nullptr);
    // layers 1..3: ping-pong bf16 sequences
    lstm_layer_kernel<HID, false><<<grid, blk, 0, stream>>>(
        nullptr, seqA, W_ih + 0 * WL, W_hh + 0 * WL, bseq + 0 * GATES, seqB, nullptr);
    lstm_layer_kernel<HID, false><<<grid, blk, 0, stream>>>(
        nullptr, seqB, W_ih + 1 * WL, W_hh + 1 * WL, bseq + 1 * GATES, seqA, nullptr);
    lstm_layer_kernel<HID, false><<<grid, blk, 0, stream>>>(
        nullptr, seqA, W_ih + 2 * WL, W_hh + 2 * WL, bseq + 2 * GATES, seqB, nullptr);
    // layer 4: only last hidden state is needed downstream
    lstm_layer_kernel<HID, false><<<grid, blk, 0, stream>>>(
        nullptr, seqB, W_ih + 3 * WL, W_hh + 3 * WL, bseq + 3 * GATES, nullptr, hlast);

    // head
    bn_kernel<<<(BATCH * HID + 255) / 256, 256, 0, stream>>>(hlast, gamma, beta,
                                                             rmean, rvar, xn);
    fc_kernel<<<(BATCH * HID + 255) / 256, 256, 0, stream>>>(xn, W1, b1, y1, HID, HID, 1);
    fc_kernel<<<(BATCH * HID + 255) / 256, 256, 0, stream>>>(y1, W2, b2, y2, HID, HID, 1);
    fc_kernel<<<(BATCH * 3 + 255) / 256, 256, 0, stream>>>(y2, W3, b3, (float*)d_out,
                                                           HID, 3, 0);
}
